// NETWORK_OF_DANS_and_VECS_16338055594192
// MI455X (gfx1250) — matrix operand order repaired and checked by reading, not compiled, not run on hardware
//
#include <hip/hip_runtime.h>
#include <math.h>

typedef __attribute__((ext_vector_type(16))) _Float16 v16h;
typedef __attribute__((ext_vector_type(8)))  float    v8f;
typedef __attribute__((ext_vector_type(4)))  float    v4f;   // native vector: OK for NT builtins

#define NN      512
#define CC      16
#define LTRANS  3
#define AGGIN   1028   // N + B = 512 + 516

// Non-temporal helpers: single-use weight streams (292 MB > 192 MB L2)
// should not displace reused lines. Lowers to global_load_* th:TH_LOAD_NT.
__device__ __forceinline__ v4f ldnt4(const float* p) {
    return __builtin_nontemporal_load((const v4f*)p);
}
__device__ __forceinline__ float ldnt(const float* p) {
    return __builtin_nontemporal_load(p);
}

// ---------------------------------------------------------------------------
// Edge kernel: one thread per edge (q,p). Streams 348 B/edge of per-edge MLP
// weights (b128 NT loads).
//   filtered[c] = tanh(a[p]*vW[c] + vb[c])                    c = 0..15
//   h1[o]       = tanh(sum_c sW1[o][c]*filtered[c] + sb1[o])  o = 0..2
//   s[q,p]      = tanh(sum_o sW2[o]*h1[o] + sb2)
// ---------------------------------------------------------------------------
__global__ __launch_bounds__(256)
void edge_kernel(const float* __restrict__ a,
                 const float* __restrict__ vW,
                 const float* __restrict__ vb,
                 const float* __restrict__ sW1,
                 const float* __restrict__ sb1,
                 const float* __restrict__ sW2,
                 const float* __restrict__ sb2,
                 float* __restrict__ s_out)
{
    const int tid = blockIdx.x * 256 + threadIdx.x;   // edge id = q*512 + p
    const int p   = tid & (NN - 1);
    const float ap = a[p];                            // reused by all q: RT

    const float* vWp = vW + (size_t)tid * CC;
    const float* vbp = vb + (size_t)tid * CC;
    v4f f[4];
#pragma unroll
    for (int i = 0; i < 4; ++i) {
        v4f w = ldnt4(vWp + 4 * i), b = ldnt4(vbp + 4 * i);
        f[i].x = tanhf(ap * w.x + b.x);
        f[i].y = tanhf(ap * w.y + b.y);
        f[i].z = tanhf(ap * w.z + b.z);
        f[i].w = tanhf(ap * w.w + b.w);
    }

    const float* sW1p = sW1 + (size_t)tid * 48;
    float h1[3];
#pragma unroll
    for (int o = 0; o < 3; ++o) {
        float acc = ldnt(sb1 + (size_t)tid * 3 + o);
#pragma unroll
        for (int i = 0; i < 4; ++i) {
            v4f w = ldnt4(sW1p + (o * 4 + i) * 4);
            acc += w.x * f[i].x + w.y * f[i].y + w.z * f[i].z + w.w * f[i].w;
        }
        h1[o] = tanhf(acc);
    }

    float sacc = ldnt(sb2 + tid);
#pragma unroll
    for (int o = 0; o < 3; ++o)
        sacc += ldnt(sW2 + (size_t)tid * 3 + o) * h1[o];
    s_out[tid] = tanhf(sacc);   // RT store: consumed by agg kernel via L2
}

// ---------------------------------------------------------------------------
// Aggregator kernel: one wave32 per node q. The 3x512 dot product
// h_pre[k] = sum_p aW1[q][k][p] * s[q][p] runs on v_wmma_f32_16x16x32_f16:
//   A rows m=0..2 hold the three weight rows (16-bit A 16x32 lane layout:
//     lanes 0-15: V0..3 -> K=0..7, V4..7 -> K=16..23;
//     lanes 16-31: V0..3 -> K=8..15, V4..7 -> K=24..31), rows m>=3 zero.
//   B holds the s-chunk replicated into all 16 columns (lane-group uniform:
//     lanes 0-15 carry K=0..7,16..23, lanes 16-31 carry K=8..15,24..31).
// f32 accumulation over 16 chunks of K=32. D layout: VGPR r, lanes 0-15 =>
// (M=r, N=lane); all columns identical, so lane 0 reads d[k] = acc[k].
// One-hot tail: layer code -> aW1[q][k][512+l+1]; node id -> aW1[q][k][516+q].
// ---------------------------------------------------------------------------
__global__ __launch_bounds__(256)
void agg_kernel(const float* __restrict__ s_buf,
                const float* __restrict__ aW1,
                const float* __restrict__ ab1,
                const float* __restrict__ aW2,
                const float* __restrict__ ab2,
                int lp1,                       // l + 1 (layer-code hot index)
                float* __restrict__ a_next,
                float* __restrict__ out_rev)   // non-null on last layer
{
    const int lane = threadIdx.x & 31;
    const int q    = blockIdx.x * 8 + (threadIdx.x >> 5);
    const int m    = lane & 15;
    const bool hi  = lane >= 16;

    const float* w  = aW1 + (size_t)q * 3 * AGGIN;   // row k at w + k*AGGIN
    const float* sv = s_buf + (size_t)q * NN;

    v8f acc;
#pragma unroll
    for (int j = 0; j < 8; ++j) acc[j] = 0.0f;

    v16h A;                       // lanes m>=3 stay zero for all iterations
#pragma unroll
    for (int j = 0; j < 16; ++j) A[j] = (_Float16)0.0f;

    for (int cb = 0; cb < NN; cb += 32) {
        if (m < 3) {
            const float* wr = w + (size_t)m * AGGIN + cb + (hi ? 8 : 0);
#pragma unroll
            for (int j = 0; j < 8; ++j) A[j]     = (_Float16)ldnt(wr + j);
#pragma unroll
            for (int j = 0; j < 8; ++j) A[8 + j] = (_Float16)ldnt(wr + 16 + j);
        }
        const float* sp = sv + cb + (hi ? 8 : 0);   // s: just-written, RT
        v16h Bv;
#pragma unroll
        for (int j = 0; j < 8; ++j) { Bv[j] = (_Float16)sp[j]; Bv[8 + j] = (_Float16)sp[16 + j]; }

        acc = __builtin_amdgcn_wmma_f32_16x16x32_f16(
                  /*neg_a=*/false, A, /*neg_b=*/false, Bv,
                  /*c_mod=*/(short)0, acc, /*reuse_a=*/false, /*reuse_b=*/false);
    }

    if (lane == 0) {
        float v = ab2[q];
#pragma unroll
        for (int k = 0; k < 3; ++k) {
            float hp = acc[k]
                     + w[(size_t)k * AGGIN + NN + lp1]       // layer one-hot
                     + w[(size_t)k * AGGIN + NN + 4 + q]     // node one-hot
                     + ab1[(size_t)q * 3 + k];
            v += aW2[(size_t)q * 3 + k] * tanhf(hp);
        }
        a_next[q] = v;
        if (out_rev) out_rev[NN - 1 - q] = v;   // reference reverses nodes
    }
}

// ---------------------------------------------------------------------------
extern "C" void kernel_launch(void* const* d_in, const int* in_sizes, int n_in,
                              void* d_out, int out_size, void* d_ws, size_t ws_size,
                              hipStream_t stream)
{
    (void)in_sizes; (void)n_in; (void)out_size; (void)ws_size;

    const float* x      = (const float*)d_in[0];
    const float* vec_W  = (const float*)d_in[1];
    const float* vec_b  = (const float*)d_in[2];
    const float* syn_W1 = (const float*)d_in[3];
    const float* syn_b1 = (const float*)d_in[4];
    const float* syn_W2 = (const float*)d_in[5];
    const float* syn_b2 = (const float*)d_in[6];
    const float* agg_W1 = (const float*)d_in[7];
    const float* agg_b1 = (const float*)d_in[8];
    const float* agg_W2 = (const float*)d_in[9];
    const float* agg_b2 = (const float*)d_in[10];

    float* s_buf = (float*)d_ws;                 // 512*512 f32 = 1 MB
    float* a0    = s_buf + (size_t)NN * NN;      // ping
    float* a1    = a0 + NN;                      // pong

    const size_t E = (size_t)NN * NN;            // edges per layer transition
    const float* a_cur = x;

    for (int l = 0; l < LTRANS; ++l) {
        edge_kernel<<<dim3((unsigned)(E / 256)), dim3(256), 0, stream>>>(
            a_cur,
            vec_W  + (size_t)l * E * CC,
            vec_b  + (size_t)l * E * CC,
            syn_W1 + (size_t)l * E * 48,
            syn_b1 + (size_t)l * E * 3,
            syn_W2 + (size_t)l * E * 3,
            syn_b2 + (size_t)l * E,
            s_buf);

        float* a_next = (l & 1) ? a1 : a0;
        agg_kernel<<<dim3(NN / 8), dim3(256), 0, stream>>>(
            s_buf,
            agg_W1 + (size_t)l * NN * 3 * AGGIN,
            agg_b1 + (size_t)l * NN * 3,
            agg_W2 + (size_t)l * NN * 3,
            agg_b2 + (size_t)l * NN,
            l + 1,
            a_next,
            (l == LTRANS - 1) ? (float*)d_out : (float*)nullptr);

        a_cur = a_next;
    }
}
